// SelfAttention_76673756168687
// MI455X (gfx1250) — matrix operand order repaired and checked by reading, not compiled, not run on hardware
//
#include <hip/hip_runtime.h>
#include <hip/hip_bf16.h>
#include <math.h>

// ---------------------------------------------------------------------------
// Problem constants (B=2, S=2048, K=1024, H=16, D=64)
// ---------------------------------------------------------------------------
#define BATCH 2
#define SEQ   2048
#define EMB   1024
#define HEADS 16
#define HDIM  64
#define ROWS  (BATCH * SEQ)          // 4096
#define SM_SCALE 0.03125f            // 1/sqrt(1024)

typedef __bf16 bf16;
typedef __attribute__((ext_vector_type(16))) __bf16 v16bf;
typedef __attribute__((ext_vector_type(8)))  __bf16 v8bf;
typedef __attribute__((ext_vector_type(8)))  float  v8f;

union AFrag { v16bf v; v8bf h[2]; };

__device__ __forceinline__ bf16 f2bf(float f) {
    union { float f; unsigned u; } a; a.f = f;
    unsigned r = (a.u + 0x7FFFu + ((a.u >> 16) & 1u)) >> 16;
    union { unsigned short s; bf16 b; } o; o.s = (unsigned short)r;
    return o.b;
}

// ---------------------------------------------------------------------------
// fp32 -> bf16 conversion
// ---------------------------------------------------------------------------
__global__ void cvt_f32_bf16(const float* __restrict__ in, bf16* __restrict__ out, int n) {
    int i = blockIdx.x * blockDim.x + threadIdx.x;
    if (i < n) out[i] = f2bf(in[i]);
}

// ---------------------------------------------------------------------------
// GEMM: C[M,N] = A[M,Kd] (bf16, row-major) x W[N,Kd]^T (bf16, row-major)
// Each wave computes a 32x64 tile. Ping-pong double-buffered k-loop
// (unrolled by two 32-wide k-stages with disjoint register buffers): no
// rotation copies, loads for one buffer overlap WMMAs on the other.
// Requires Kd % 64 == 0 (true here: 1024).
// __launch_bounds__(128, 4): ~256 VGPR/wave budget -> no scratch spills,
// 4 waves/SIMD for latency hiding.
// mode 0: store bf16 row-major [M,N]
// mode 1: store bf16 per-head transposed  Vt[b][h][d][s]   (for attention V)
// mode 2: store f32 row-major + bias      (final output projection)
// ---------------------------------------------------------------------------
__global__ __launch_bounds__(128, 4)
void gemm_bf16_nt(const bf16* __restrict__ A, const bf16* __restrict__ W,
                  void* __restrict__ out, const float* __restrict__ bias,
                  int M, int N, int Kd, int mode) {
    const int wid  = threadIdx.x >> 5;
    const int lane = threadIdx.x & 31;
    const int widx = blockIdx.x * (blockDim.x >> 5) + wid;
    const int ntiles = N >> 6;
    if (widx >= (M >> 5) * ntiles) return;
    const int m0 = (widx / ntiles) << 5;
    const int n0 = (widx % ntiles) << 6;

    const int am  = lane & 15;
    const int a8  = (lane >> 4) << 3;   // acc rows: +8 row shift for upper half-wave
    const int b16 = (lane >> 4) << 4;   // B frag: +16 elem shift for upper half-wave

    // loop-invariant per-lane row pointers
    const bf16* pa[2];
#pragma unroll
    for (int hlf = 0; hlf < 2; ++hlf)
        pa[hlf] = A + (size_t)(m0 + hlf * 16 + am) * Kd + b16;
    const bf16* pw[4];
#pragma unroll
    for (int t = 0; t < 4; ++t)
        pw[t] = W + (size_t)(n0 + t * 16 + am) * Kd + b16;

    v8f acc[2][4] = {};

#define LOAD_STAGE(aa, bb, kof)                                          \
    do {                                                                 \
        _Pragma("unroll")                                                \
        for (int hlf = 0; hlf < 2; ++hlf) {                              \
            (aa)[hlf].h[0] = *(const v8bf*)(pa[hlf] + (kof));            \
            (aa)[hlf].h[1] = *(const v8bf*)(pa[hlf] + (kof) + 8);        \
        }                                                                \
        _Pragma("unroll")                                                \
        for (int t = 0; t < 4; ++t)                                      \
            (bb)[t] = *(const v16bf*)(pw[t] + (kof));                    \
    } while (0)

#define DO_WMMA(aa, bb)                                                  \
    do {                                                                 \
        _Pragma("unroll")                                                \
        for (int t = 0; t < 4; ++t) {                                    \
            _Pragma("unroll")                                            \
            for (int hlf = 0; hlf < 2; ++hlf) {                          \
                acc[hlf][t] = __builtin_amdgcn_wmma_f32_16x16x32_bf16(   \
                    false, (aa)[hlf].v, false, (bb)[t], (short)0,        \
                    acc[hlf][t], false, false);                          \
            }                                                            \
        }                                                                \
    } while (0)

    AFrag aA[2], aB[2]; v16bf bA[4], bB[4];
    LOAD_STAGE(aA, bA, 0);
    for (int k0 = 0; k0 < Kd - 64; k0 += 64) {
        LOAD_STAGE(aB, bB, k0 + 32);   // fill B while computing A
        DO_WMMA(aA, bA);
        LOAD_STAGE(aA, bA, k0 + 64);   // fill A while computing B
        DO_WMMA(aB, bB);
    }
    LOAD_STAGE(aB, bB, Kd - 32);       // last stage
    DO_WMMA(aA, bA);
    DO_WMMA(aB, bB);

#undef LOAD_STAGE
#undef DO_WMMA

#pragma unroll
    for (int hlf = 0; hlf < 2; ++hlf) {
#pragma unroll
        for (int t = 0; t < 4; ++t) {
#pragma unroll
            for (int r = 0; r < 8; ++r) {
                const int mg = m0 + hlf * 16 + r + a8;
                const int ng = n0 + t * 16 + am;
                const float v = acc[hlf][t][r];
                if (mode == 0) {
                    ((bf16*)out)[(size_t)mg * N + ng] = f2bf(v);
                } else if (mode == 1) {
                    const int bb = mg >> 11, ss = mg & 2047;
                    const int hh = ng >> 6,  dd = ng & 63;
                    ((bf16*)out)[((size_t)((bb * HEADS + hh) * HDIM + dd)) * SEQ + ss] = f2bf(v);
                } else {
                    ((float*)out)[(size_t)mg * N + ng] = v + bias[ng];
                }
            }
        }
    }
}

// ---------------------------------------------------------------------------
// Flash attention: one wave per (b, h, 16-query tile). Key blocks of 32.
// Q,K bf16 row-major [ROWS, EMB]; V transposed bf16 [B][H][D][S]; O bf16 [ROWS, EMB].
// ---------------------------------------------------------------------------
__global__ __launch_bounds__(128, 4)
void flash_attn(const bf16* __restrict__ Q, const bf16* __restrict__ Km,
                const bf16* __restrict__ Vt, bf16* __restrict__ O) {
    const int wid  = threadIdx.x >> 5;
    const int lane = threadIdx.x & 31;
    const int w    = blockIdx.x * (blockDim.x >> 5) + wid;
    if (w >= BATCH * HEADS * (SEQ / 16)) return;
    const int qt = w & 127;
    const int bh = w >> 7;
    const int h  = bh & (HEADS - 1);
    const int b  = bh >> 4;
    const int q0 = qt << 4;

    const int am  = lane & 15;
    const int a8  = (lane >> 4) << 3;
    const int b16 = (lane >> 4) << 4;

    __shared__ __align__(32) bf16 smem[4][16 * 32];
    bf16* P = smem[wid];

    // Q fragments (d = 0..31 and 32..63)
    AFrag aq[2];
#pragma unroll
    for (int d2 = 0; d2 < 2; ++d2) {
        const bf16* qrow = Q + (size_t)(b * SEQ + q0 + am) * EMB + h * HDIM + d2 * 32 + b16;
        aq[d2].h[0] = *(const v8bf*)(qrow);
        aq[d2].h[1] = *(const v8bf*)(qrow + 8);
    }

    float mrow[8], lrow[8];
#pragma unroll
    for (int r = 0; r < 8; ++r) { mrow[r] = -__builtin_inff(); lrow[r] = 0.0f; }
    v8f oacc[4] = {};

    for (int j0 = 0; j0 < q0 + 16; j0 += 32) {
        // ---- scores: S(16x32) = Q(16x64) . K^T ----
        v8f s[2] = {};
#pragma unroll
        for (int d2 = 0; d2 < 2; ++d2) {
#pragma unroll
            for (int t = 0; t < 2; ++t) {
                const bf16* krow = Km + (size_t)(b * SEQ + j0 + t * 16 + am) * EMB
                                      + h * HDIM + d2 * 32 + b16;
                v16bf bk = *(const v16bf*)krow;
                s[t] = __builtin_amdgcn_wmma_f32_16x16x32_bf16(
                    false, aq[d2].v, false, bk, (short)0, s[t], false, false);
            }
        }

        // ---- online softmax over the 32-key block ----
#pragma unroll
        for (int r = 0; r < 8; ++r) {
            const int mg = q0 + r + a8;                // global query row of this slot
            float v0 = s[0][r] * SM_SCALE;
            float v1 = s[1][r] * SM_SCALE;
            if (j0 + am      > mg) v0 = -__builtin_inff();
            if (j0 + 16 + am > mg) v1 = -__builtin_inff();
            float mx = fmaxf(v0, v1);
#pragma unroll
            for (int off = 1; off < 16; off <<= 1)
                mx = fmaxf(mx, __shfl_xor(mx, off, 32));
            const float mnew  = fmaxf(mrow[r], mx);
            const float alpha = __expf(mrow[r] - mnew);
            const float p0 = __expf(v0 - mnew);
            const float p1 = __expf(v1 - mnew);
            float sum = p0 + p1;
#pragma unroll
            for (int off = 1; off < 16; off <<= 1)
                sum += __shfl_xor(sum, off, 32);
            lrow[r] = lrow[r] * alpha + sum;
            mrow[r] = mnew;
#pragma unroll
            for (int t2 = 0; t2 < 4; ++t2) oacc[t2][r] *= alpha;
            // stage P (C-layout -> LDS row-major 16x32)
            P[(r + a8) * 32 + am]      = f2bf(p0);
            P[(r + a8) * 32 + 16 + am] = f2bf(p1);
        }

        // ---- reload P as A fragment (16x32) ----
        AFrag ap;
        ap.h[0] = *(const v8bf*)&P[am * 32 + b16];
        ap.h[1] = *(const v8bf*)&P[am * 32 + b16 + 8];

        // ---- O += P . V  (V transposed: contiguous along s) ----
#pragma unroll
        for (int t2 = 0; t2 < 4; ++t2) {
            const bf16* vrow = Vt + (size_t)((b * HEADS + h) * HDIM + t2 * 16 + am) * SEQ
                                  + j0 + b16;
            v16bf bv = *(const v16bf*)vrow;
            oacc[t2] = __builtin_amdgcn_wmma_f32_16x16x32_bf16(
                false, ap.v, false, bv, (short)0, oacc[t2], false, false);
        }
    }

    // ---- normalize and store O (bf16 row-major [ROWS, EMB]) ----
#pragma unroll
    for (int t2 = 0; t2 < 4; ++t2) {
#pragma unroll
        for (int r = 0; r < 8; ++r) {
            const float o = oacc[t2][r] / lrow[r];
            O[(size_t)(b * SEQ + q0 + r + a8) * EMB + h * HDIM + t2 * 16 + am] = f2bf(o);
        }
    }
}

// ---------------------------------------------------------------------------
// Launch
// ---------------------------------------------------------------------------
extern "C" void kernel_launch(void* const* d_in, const int* in_sizes, int n_in,
                              void* d_out, int out_size, void* d_ws, size_t ws_size,
                              hipStream_t stream) {
    const float* x  = (const float*)d_in[0];
    const float* Wk = (const float*)d_in[1];
    const float* Wq = (const float*)d_in[2];
    const float* Wv = (const float*)d_in[3];
    const float* Wu = (const float*)d_in[4];
    const float* bu = (const float*)d_in[5];

    char* ws = (char*)d_ws;
    const size_t MB = 1024u * 1024u;
    bf16* xb  = (bf16*)(ws + 0);        // 8 MB : x        bf16 [4096,1024]
    bf16* wqb = (bf16*)(ws + 8  * MB);  // 2 MB
    bf16* wkb = (bf16*)(ws + 10 * MB);  // 2 MB
    bf16* wvb = (bf16*)(ws + 12 * MB);  // 2 MB
    bf16* wub = (bf16*)(ws + 14 * MB);  // 2 MB
    bf16* Qb  = (bf16*)(ws + 16 * MB);  // 8 MB
    bf16* Kb  = (bf16*)(ws + 24 * MB);  // 8 MB
    bf16* Vt  = (bf16*)(ws + 32 * MB);  // 8 MB : V^T [B][H][D][S]
    bf16* Ob  = (bf16*)(ws + 40 * MB);  // 8 MB

    const int NX = ROWS * EMB;          // 4194304
    const int NW = EMB * EMB;           // 1048576
    cvt_f32_bf16<<<(NX + 255) / 256, 256, 0, stream>>>(x,  xb,  NX);
    cvt_f32_bf16<<<(NW + 255) / 256, 256, 0, stream>>>(Wq, wqb, NW);
    cvt_f32_bf16<<<(NW + 255) / 256, 256, 0, stream>>>(Wk, wkb, NW);
    cvt_f32_bf16<<<(NW + 255) / 256, 256, 0, stream>>>(Wv, wvb, NW);
    cvt_f32_bf16<<<(NW + 255) / 256, 256, 0, stream>>>(Wu, wub, NW);

    // waves = (4096/32)*(1024/64) = 2048 ; 4 waves per 128-thread block
    const int gemm_blocks = ((ROWS / 32) * (EMB / 64)) / 4;
    gemm_bf16_nt<<<gemm_blocks, 128, 0, stream>>>(xb, wqb, Qb, nullptr, ROWS, EMB, EMB, 0);
    gemm_bf16_nt<<<gemm_blocks, 128, 0, stream>>>(xb, wkb, Kb, nullptr, ROWS, EMB, EMB, 0);
    gemm_bf16_nt<<<gemm_blocks, 128, 0, stream>>>(xb, wvb, Vt, nullptr, ROWS, EMB, EMB, 1);

    // waves = B*H*(S/16) = 4096
    const int fa_blocks = (BATCH * HEADS * (SEQ / 16)) / 4;
    flash_attn<<<fa_blocks, 128, 0, stream>>>(Qb, Kb, Vt, Ob);

    gemm_bf16_nt<<<gemm_blocks, 128, 0, stream>>>(Ob, wub, d_out, bu, ROWS, EMB, EMB, 2);
}
